// GraphSAGE_34986803593269
// MI455X (gfx1250) — matrix operand order repaired and checked by reading, not compiled, not run on hardware
//
#include <hip/hip_runtime.h>

typedef __attribute__((ext_vector_type(16))) __bf16 v16bf;
typedef __attribute__((ext_vector_type(8)))  __bf16 v8bf;
typedef __attribute__((ext_vector_type(8)))  float  v8f;

#define N_NODES 100000
#define D_FEAT  128
#define K_NEIGH 16
#define F_OUT   128
#define TWOD    256
#define LDS_ROW 264   // 256 bf16 + 8 pad to stagger LDS banks across rows

// ---------------------------------------------------------------------------
// Prep: transpose W[l][k][n] -> Wt[l][n][k] and split fp32 into bf16 hi + lo
// so that x = hi + lo with |lo| ~ 2^-9 |x|; GEMM uses hi*hi + hi*lo + lo*hi.
// ---------------------------------------------------------------------------
__global__ __launch_bounds__(256) void prep_weights_kernel(
    const float* __restrict__ Ws, __bf16* __restrict__ wt_hi,
    __bf16* __restrict__ wt_lo) {
  int idx = blockIdx.x * 256 + threadIdx.x;   // enumerates [l][k][n], 65536 elems
  int l = idx >> 15;
  int k = (idx >> 7) & 255;
  int n = idx & 127;
  float w = Ws[idx];
  __bf16 hi = (__bf16)w;
  __bf16 lo = (__bf16)(w - (float)hi);
  int o = (l << 15) + n * TWOD + k;           // [l][n][k]
  wt_hi[o] = hi;
  wt_lo[o] = lo;
}

// ---------------------------------------------------------------------------
// Fused GraphSAGE layer: gather 16 neighbors/node, mean, concat with self,
// then cat[16x256] @ W[256x128] + b via bf16 split WMMA. One block = 16 nodes.
// ---------------------------------------------------------------------------
__global__ __launch_bounds__(256) void sage_layer_kernel(
    const float*  __restrict__ hin,    // [N,128] layer input
    const int*    __restrict__ nidx,   // [N,16]  this layer's neighbor ids
    const __bf16* __restrict__ wt_hi,  // [128,256] transposed hi weights
    const __bf16* __restrict__ wt_lo,  // [128,256] transposed lo weights
    const float*  __restrict__ bias,   // [128]
    float*        __restrict__ out,    // [N,128]
    const int     apply_relu) {
  __shared__ __bf16 s_hi[16 * LDS_ROW];
  __shared__ __bf16 s_lo[16 * LDS_ROW];
  __shared__ int    s_idx[16 * 16];

  const int t    = threadIdx.x;
  const int tile = blockIdx.x;               // 16 nodes per tile, N % 16 == 0

  // Stage this tile's neighbor indices (one per thread).
  {
    int r = t >> 4, j = t & 15;
    s_idx[t] = nidx[(size_t)(tile * 16 + r) * K_NEIGH + j];
  }
  __syncthreads();

  // ---- Phase 1: build cat = [self | mean(neigh)] as bf16 hi/lo in LDS ----
  {
    const int r    = t >> 4;                 // node row in tile
    const int c8   = (t & 15) * 8;           // 8 feature columns per thread
    const int node = tile * 16 + r;
    __bf16* hrow = s_hi + r * LDS_ROW;
    __bf16* lrow = s_lo + r * LDS_ROW;

    // self half (k = 0..127)
    const float* selfp = hin + (size_t)node * D_FEAT + c8;
    #pragma unroll
    for (int q = 0; q < 8; ++q) {
      float v = selfp[q];
      __bf16 h = (__bf16)v;
      hrow[c8 + q] = h;
      lrow[c8 + q] = (__bf16)(v - (float)h);
    }

    // mean-of-neighbors half (k = 128..255): L2-resident gather
    float acc[8];
    #pragma unroll
    for (int q = 0; q < 8; ++q) acc[q] = 0.0f;
    #pragma unroll 4
    for (int j = 0; j < K_NEIGH; ++j) {
      const float* p = hin + (size_t)s_idx[(r << 4) + j] * D_FEAT + c8;
      #pragma unroll
      for (int q = 0; q < 8; ++q) acc[q] += p[q];
    }
    #pragma unroll
    for (int q = 0; q < 8; ++q) {
      float m = acc[q] * (1.0f / 16.0f);
      __bf16 h = (__bf16)m;
      hrow[D_FEAT + c8 + q] = h;
      lrow[D_FEAT + c8 + q] = (__bf16)(m - (float)h);
    }
  }
  __syncthreads();

  // ---- Phase 2: per-wave 16x16 output tile via v_wmma_f32_16x16x32_bf16 ----
  const int wave = t >> 5;                   // 8 waves -> 8 column tiles
  const int lane = t & 31;                   // wave32
  const int m    = lane & 15;                // A row (M) and B col (N) index
  const int g    = lane >> 4;                // lane half selects K sub-range

  const __bf16* ah_row  = s_hi + m * LDS_ROW;
  const __bf16* al_row  = s_lo + m * LDS_ROW;
  const __bf16* bh_base = wt_hi + (size_t)(wave * 16 + m) * TWOD;
  const __bf16* bl_base = wt_lo + (size_t)(wave * 16 + m) * TWOD;

  v8f acc = {};
  #pragma unroll
  for (int kk = 0; kk < TWOD; kk += 32) {
    v16bf ahi, alo, bhi, blo;
    // A 16x32 bf16: elems 0..7 <- K=kk+g*8.., elems 8..15 <- K=kk+16+g*8..
    ((v8bf*)&ahi)[0] = *(const v8bf*)(ah_row + kk + g * 8);
    ((v8bf*)&ahi)[1] = *(const v8bf*)(ah_row + kk + g * 8 + 16);
    ((v8bf*)&alo)[0] = *(const v8bf*)(al_row + kk + g * 8);
    ((v8bf*)&alo)[1] = *(const v8bf*)(al_row + kk + g * 8 + 16);
    // B 32x16 bf16: elems 0..7 <- K=kk+g*8.., elems 8..15 <- K=kk+16+g*8..
    ((v8bf*)&bhi)[0] = *(const v8bf*)(bh_base + kk + g * 8);
    ((v8bf*)&bhi)[1] = *(const v8bf*)(bh_base + kk + g * 8 + 16);
    ((v8bf*)&blo)[0] = *(const v8bf*)(bl_base + kk + g * 8);
    ((v8bf*)&blo)[1] = *(const v8bf*)(bl_base + kk + g * 8 + 16);
    // fp32-accurate split product: hi*hi + hi*lo + lo*hi
    acc = __builtin_amdgcn_wmma_f32_16x16x32_bf16(false, ahi, false, bhi,
                                                  (short)0, acc, false, false);
    acc = __builtin_amdgcn_wmma_f32_16x16x32_bf16(false, ahi, false, blo,
                                                  (short)0, acc, false, false);
    acc = __builtin_amdgcn_wmma_f32_16x16x32_bf16(false, alo, false, bhi,
                                                  (short)0, acc, false, false);
  }

  // D layout: VGPR r -> M = g*8 + r, lane%16 -> N. Fuse bias (+ optional relu).
  const int col = wave * 16 + m;
  const float b = bias[col];
  #pragma unroll
  for (int r = 0; r < 8; ++r) {
    int row = tile * 16 + g * 8 + r;
    float v = acc[r] + b;
    if (apply_relu) v = v > 0.0f ? v : 0.0f;
    out[(size_t)row * F_OUT + col] = v;
  }
}

// ---------------------------------------------------------------------------
extern "C" void kernel_launch(void* const* d_in, const int* in_sizes, int n_in,
                              void* d_out, int out_size, void* d_ws, size_t ws_size,
                              hipStream_t stream) {
  const float* feats = (const float*)d_in[0];           // [100000,128]
  const int*   neigh = (const int*)d_in[1];             // [2,100000,16]
  const float* Ws    = (const float*)d_in[2];           // [2,256,128]
  const float* bs    = (const float*)d_in[3];           // [2,128]
  float*       out   = (float*)d_out;                   // [100000,128]

  char* ws = (char*)d_ws;
  float*  h1    = (float*)ws;                           // 51.2 MB: relu(z1)
  __bf16* wt_hi = (__bf16*)(ws + (size_t)N_NODES * D_FEAT * sizeof(float));
  __bf16* wt_lo = wt_hi + 2 * F_OUT * TWOD;             // [2][128][256] each

  prep_weights_kernel<<<256, 256, 0, stream>>>(Ws, wt_hi, wt_lo);

  const int blocks = N_NODES / 16;                      // 6250
  // Layer 0: feats -> h1 = relu(cat @ W0 + b0)
  sage_layer_kernel<<<blocks, 256, 0, stream>>>(
      feats, neigh, wt_hi, wt_lo, bs, h1, 1);
  // Layer 1: h1 -> z (pre-relu) = d_out
  sage_layer_kernel<<<blocks, 256, 0, stream>>>(
      h1, neigh + (size_t)N_NODES * K_NEIGH,
      wt_hi + F_OUT * TWOD, wt_lo + F_OUT * TWOD, bs + F_OUT, out, 0);
}
